// NoiseFilter_30889404793032
// MI455X (gfx1250) — matrix operand order repaired and checked by reading, not compiled, not run on hardware
//
#include <hip/hip_runtime.h>

typedef __attribute__((ext_vector_type(16))) _Float16 v16h;
typedef __attribute__((ext_vector_type(8)))  float    v8f;

#define TPB        128
#define NWAVE      4
#define RPW        16      // rows per workgroup
#define NB_        65
#define KPAD       96      // stage-A K padded to 3x32
#define IRLEN      128
#define HOP_       256
#define NPADLEN    528     // 272 leading zeros + 256 noise samples (f16)
#define ZPFX       272     // zero prefix halves
#define KLEN       112     // reversed-kernel copy length
#define NCOPY      2       // parity-shifted copies of the reversed kernel
#define NWG        4096    // 65536 rows / 16

union V16 { v16h v; uint4 q[2]; uint2 d[4]; };
union H4  { uint2 d; _Float16 h[4]; };

// ---- setup: TRANSPOSED cosine-basis * window matrix  D[128 x 96] (f16) ----
__global__ __launch_bounds__(IRLEN) void build_dmat(_Float16* __restrict__ D) {
  int k = blockIdx.x;      // 0..95 (basis index, zero-padded past 65)
  int j = threadIdx.x;     // 0..127 (time index)
  float v = 0.0f;
  if (k < NB_) {
    const float PI64 = 3.14159265358979323846f / 64.0f;
    float ck = (k == 0 || k == 64) ? (1.0f / 128.0f) : (2.0f / 128.0f);
    float w  = 0.5f * (1.0f + __cosf(PI64 * (float)j));   // ½(1+cos(πj/64))
    int m = (k * j) & 127;                                // exact range reduction
    v = ck * __cosf(PI64 * (float)m) * w;
  }
  D[j * KPAD + k] = (_Float16)v;                          // transposed store
}

__global__ __launch_bounds__(TPB) void noise_filter(
    const float* __restrict__ fb, const float* __restrict__ nz,
    const _Float16* __restrict__ Dg, float* __restrict__ out) {
  __shared__ __align__(16) _Float16 sAmp[RPW][KPAD];              // 3 KB
  __shared__ __align__(16) _Float16 sK[RPW][2][NCOPY][KLEN];      // 14 KB [row][band][shift][x]
  __shared__ __align__(32) _Float16 sNp[NWAVE][2][NPADLEN];       // 8.25 KB double-buffered

  const int tid  = threadIdx.x;
  const int lane = tid & 31;
  const int wave = tid >> 5;
  const int half = lane >> 4;
  const int l16  = lane & 15;
  const long row0 = (long)blockIdx.x * RPW;

  // ---- phase 0: zero kernel copies + noise zero-prefixes; stage amp (f32->f16) ----
  {
    uint32_t* z = (uint32_t*)&sK[0][0][0][0];
    for (int i = tid; i < RPW * 2 * NCOPY * KLEN / 2; i += TPB) z[i] = 0u;
  }
  for (int i = tid; i < NWAVE * 2 * (ZPFX / 2); i += TPB) {
    int w = i / (2 * (ZPFX / 2));
    int b = (i / (ZPFX / 2)) & 1;
    int d = i % (ZPFX / 2);
    *((uint32_t*)&sNp[w][b][0] + d) = 0u;
  }
  for (int i = tid; i < RPW * KPAD; i += TPB) {
    int r = i / KPAD, k = i % KPAD;
    float v = (k < NB_) ? fb[(row0 + r) * NB_ + k] : 0.0f;
    sAmp[r][k] = (_Float16)v;
  }
  __syncthreads();

  // ---- Stage A: h~[16x128] = amp[16x96] x D[96x128]; scatter results directly
  //      into 2 parity-shifted copies of the reversed band kernels ----
  V16 af[3];
#pragma unroll
  for (int s = 0; s < 3; ++s) {
    const int kk = s * 32;
    af[s].q[0] = *(const uint4*)&sAmp[l16][kk + (half << 3)];
    af[s].q[1] = *(const uint4*)&sAmp[l16][kk + (half << 3) + 16];
  }
#pragma unroll
  for (int t = 0; t < 2; ++t) {
    const int n0 = (wave * 2 + t) * 16;
    const _Float16* drow = Dg + (size_t)(n0 + l16) * KPAD;   // L2-resident
    v8f acc = {};
#pragma unroll
    for (int s = 0; s < 3; ++s) {
      const int kk = s * 32;
      V16 bf;
      bf.q[0] = *(const uint4*)(drow + kk + (half << 3));
      bf.q[1] = *(const uint4*)(drow + kk + (half << 3) + 16);
      acc = __builtin_amdgcn_wmma_f32_16x16x32_f16(false, af[s].v, false, bf.v,
                                                   (short)0, acc, false, false);
    }
    // h~[M][j], j = n0 + l16.  band0 (j<64): hrevA[t]=h~[96-t]; band1: hrevB[t]=h~[160-t]
    const int band = (n0 < 64) ? 0 : 1;
    const int tb   = ((n0 < 64) ? 96 : 160) - (n0 + l16);
#pragma unroll
    for (int v = 0; v < 8; ++v) {
      const int M = v + (half << 3);
      const _Float16 val = (_Float16)acc[v];
#pragma unroll
      for (int sh = 0; sh < NCOPY; ++sh)
        sK[M][band][sh][tb - sh] = val;   // S_sh[x] = hrev[x+sh]
    }
  }
  __syncthreads();

  // ---- Stage B: per row, Out'[pi][pb] = sum_k hrev[16+k-pi] * npad[16pb+k+off] ----
  const int s_ = l16 & 1;                 // lane-constant parity copy (= j0 & 1)
#pragma unroll
  for (int i = 0; i < 4; ++i) {
    const int lr = wave * 4 + i;
    const long grow = row0 + lr;
    const int buf = i & 1;
    // stage noise: 64 quads of (2u-1) in f16 behind the static zero prefix
#pragma unroll
    for (int m = 0; m < 2; ++m) {
      const int q = lane + 32 * m;
      float4 nn = *(const float4*)&nz[grow * HOP_ + 4 * q];
      H4 cv;
      cv.h[0] = (_Float16)(nn.x * 2.0f - 1.0f);
      cv.h[1] = (_Float16)(nn.y * 2.0f - 1.0f);
      cv.h[2] = (_Float16)(nn.z * 2.0f - 1.0f);
      cv.h[3] = (_Float16)(nn.w * 2.0f - 1.0f);
      *(uint2*)&sNp[wave][buf][ZPFX + 4 * q] = cv.d;
    }
    v8f acc = {};
#pragma unroll
    for (int band = 0; band < 2; ++band) {
      const int noff = band ? 0 : 192;
      const _Float16* kb = &sK[lr][band][s_][0];
#pragma unroll
      for (int st = 0; st < 3; ++st) {
        const int kk = st * 32;
        V16 a, b;
        const int x0 = 16 + kk + (half << 3) - l16 - s_;     // even
        a.d[0] = *(const uint2*)(kb + x0);
        a.d[1] = *(const uint2*)(kb + x0 + 4);
        a.d[2] = *(const uint2*)(kb + x0 + 16);
        a.d[3] = *(const uint2*)(kb + x0 + 20);
        const int i0 = (l16 << 4) + kk + (half << 3) + noff;  // multiple of 8
        b.q[0] = *(const uint4*)&sNp[wave][buf][i0];
        b.q[1] = *(const uint4*)&sNp[wave][buf][i0 + 16];
        acc = __builtin_amdgcn_wmma_f32_16x16x32_f16(false, a.v, false, b.v,
                                                     (short)0, acc, false, false);
      }
    }
    // Out'[M=pi][N=pb] -> p = 16*l16 + v + 8*half : two 16B stores per lane
    float* op = &out[grow * HOP_ + (l16 << 4) + (half << 3)];
    float4 o0; o0.x = acc[0]; o0.y = acc[1]; o0.z = acc[2]; o0.w = acc[3];
    float4 o1; o1.x = acc[4]; o1.y = acc[5]; o1.z = acc[6]; o1.w = acc[7];
    *(float4*)op = o0;
    *(float4*)(op + 4) = o1;
  }
}

extern "C" void kernel_launch(void* const* d_in, const int* in_sizes, int n_in,
                              void* d_out, int out_size, void* d_ws, size_t ws_size,
                              hipStream_t stream) {
  const float* fb = (const float*)d_in[0];   // filter_bank [16,4096,65] f32
  const float* nz = (const float*)d_in[1];   // noise_u     [16,4096,256] f32
  float* outp = (float*)d_out;               // [16, 4096*256, 1] f32
  _Float16* D = (_Float16*)d_ws;             // 128*96*2 = 24 KB scratch (transposed)

  build_dmat<<<dim3(KPAD), dim3(IRLEN), 0, stream>>>(D);
  noise_filter<<<dim3(NWG), dim3(TPB), 0, stream>>>(fb, nz, D, outp);
}
